// Word2Vec_68152541053411
// MI455X (gfx1250) — matrix operand order repaired and checked by reading, not compiled, not run on hardware
//
#include <hip/hip_runtime.h>

// ---------------- problem constants ----------------
#define VOCAB  100000
#define DIM    128
#define BATCH  2048
#define CTX    4
#define NT     (VOCAB / 16)   // 6250 column tiles of 16
#define RT     (BATCH / 16)   // 128 row tiles of 16
#define NSPLIT 8              // N-range splits per row tile (pass1/pass2 grids = RT*NSPLIT)

typedef __bf16 v8bf  __attribute__((ext_vector_type(8)));
typedef __bf16 v16bf __attribute__((ext_vector_type(16)));
typedef float  v8f   __attribute__((ext_vector_type(8)));

// fp32 -> bf16 round-to-nearest-even
__device__ __forceinline__ unsigned short f2bf(float f) {
    unsigned int u = __float_as_uint(f);
    u += 0x7FFFu + ((u >> 16) & 1u);
    return (unsigned short)(u >> 16);
}

// ---------------------------------------------------------------------------
// Kernel 0: pack W_out (fp32, [DIM][VOCAB] row-major) into bf16 tiles laid out
// as this file's B-fragment (same k order as A): tile nt holds K=128 x N=16; chunk c (K=32)
// is 512 bf16; lane L's 16 bf16 (K = c*32 + 16*(L/16) + j, n = nt*16 + L%16)
// are contiguous 32 bytes.  25.6 MB total -> stays L2-resident.
// ---------------------------------------------------------------------------
__global__ void __launch_bounds__(128)
pack_wout(const float* __restrict__ W_out, unsigned short* __restrict__ packB) {
    const int tid = threadIdx.x;
    const int c = tid >> 5;          // K-chunk 0..3
    const int L = tid & 31;          // lane
    const int nt = blockIdx.x;       // column tile
    const int n  = nt * 16 + (L & 15);
    const int kb = c * 32 + (L >> 4) * 16;

    unsigned int q[8];
#pragma unroll
    for (int i = 0; i < 8; ++i) {
        unsigned short lo = f2bf(W_out[(size_t)(kb + 2 * i    ) * VOCAB + n]);
        unsigned short hi = f2bf(W_out[(size_t)(kb + 2 * i + 1) * VOCAB + n]);
        q[i] = (unsigned int)lo | ((unsigned int)hi << 16);
    }
    uint4* dst = (uint4*)(packB + (size_t)nt * 2048 + c * 512 + L * 16);
    dst[0] = make_uint4(q[0], q[1], q[2], q[3]);
    dst[1] = make_uint4(q[4], q[5], q[6], q[7]);
}

// ---------------------------------------------------------------------------
// Kernel 1: proj[b][d] = mean_j W_in[ctx[b][j]][d], stored as bf16.
// ---------------------------------------------------------------------------
__global__ void __launch_bounds__(128)
gather_mean(const int* __restrict__ ctx, const float* __restrict__ W_in,
            unsigned short* __restrict__ proj) {
    const int b = blockIdx.x;
    const int d = threadIdx.x;
    float s = 0.f;
#pragma unroll
    for (int j = 0; j < CTX; ++j) {
        int w = ctx[b * CTX + j];
        s += W_in[(size_t)w * DIM + d];
    }
    proj[(size_t)b * DIM + d] = f2bf(s * 0.25f);
}

// ---------------------------------------------------------------------------
// WMMA helpers
// ---------------------------------------------------------------------------
// A-fragment (16x32 bf16): lane L holds row M=L%16; K = {kb..kb+15}
// with kb = c*32 + 16*(L/16).
__device__ __forceinline__ v16bf load_a_frag(const unsigned short* __restrict__ proj,
                                             int rowbase, int L, int c) {
    const int kb = c * 32 + (L >> 4) * 16;
    const __bf16* p = (const __bf16*)(proj + (size_t)(rowbase + (L & 15)) * DIM + kb);
    v8bf h0 = *(const v8bf*)p;
    v8bf h1 = *(const v8bf*)(p + 8);
    return __builtin_shufflevector(h0, h1, 0, 1, 2, 3, 4, 5, 6, 7,
                                           8, 9, 10, 11, 12, 13, 14, 15);
}

// One 16x16 logits tile: K=128 as 4 chained v_wmma_f32_16x16x32_bf16.
// All 8 b128 loads are issued first (one clause, staggered s_wait_loadcnt),
// and the next tile (stride 32 tiles = +128 KB) is prefetched.
__device__ __forceinline__ v8f logits_tile(const v16bf a[4],
                                           const unsigned short* __restrict__ packB,
                                           int nt, int L) {
    const __bf16* base = (const __bf16*)(packB + (size_t)nt * 2048);

    v16bf b0 = *(const v16bf*)(base + 0 * 512 + L * 16);
    v16bf b1 = *(const v16bf*)(base + 1 * 512 + L * 16);
    v16bf b2 = *(const v16bf*)(base + 2 * 512 + L * 16);
    v16bf b3 = *(const v16bf*)(base + 3 * 512 + L * 16);

    // Warm next iteration's tile (speculative; OOB prefetch is dropped).
    __builtin_prefetch(base + (size_t)32 * 2048 + L * 16, 0, 3);

    v8f c = {};
    c = __builtin_amdgcn_wmma_f32_16x16x32_bf16(false, a[0], false, b0, (short)0, c, false, false);
    c = __builtin_amdgcn_wmma_f32_16x16x32_bf16(false, a[1], false, b1, (short)0, c, false, false);
    c = __builtin_amdgcn_wmma_f32_16x16x32_bf16(false, a[2], false, b2, (short)0, c, false, false);
    c = __builtin_amdgcn_wmma_f32_16x16x32_bf16(false, a[3], false, b3, (short)0, c, false, false);
    return c;
}

__device__ __forceinline__ void olmerge(float& M, float& S, float m2, float s2) {
    float mn = fmaxf(M, m2);
    S = S * __expf(M - mn) + s2 * __expf(m2 - mn);
    M = mn;
}

// ---------------------------------------------------------------------------
// Kernel 2 (pass 1): online-softmax stats.  Block = (rowtile rt, split sp),
// 4 waves; wave w covers nt = sp*4 + w, stride 32.  C-fragment layout: VGPR r
// holds (row = rowbase + 8*(L/16) + r, col = nt*16 + L%16).
// ---------------------------------------------------------------------------
__global__ void __launch_bounds__(128)
pass1_stats(const unsigned short* __restrict__ proj,
            const unsigned short* __restrict__ packB,
            float* __restrict__ partial) {
    const int tid  = threadIdx.x;
    const int wave = tid >> 5;
    const int L    = tid & 31;
    const int rt   = blockIdx.x >> 3;
    const int sp   = blockIdx.x & (NSPLIT - 1);
    const int rowbase = rt * 16;

    v16bf a[4];
#pragma unroll
    for (int c = 0; c < 4; ++c) a[c] = load_a_frag(proj, rowbase, L, c);

    float m[8], s[8];
#pragma unroll
    for (int r = 0; r < 8; ++r) { m[r] = -1e30f; s[r] = 0.f; }

    for (int nt = sp * 4 + wave; nt < NT; nt += NSPLIT * 4) {
        v8f cf = logits_tile(a, packB, nt, L);
#pragma unroll
        for (int r = 0; r < 8; ++r) {
            float v = cf[r];
            float mn = fmaxf(m[r], v);
            s[r] = s[r] * __expf(m[r] - mn) + __expf(v - mn);
            m[r] = mn;
        }
    }

    // column reduction across the 16 lanes of each wave-half
#pragma unroll
    for (int off = 1; off < 16; off <<= 1) {
#pragma unroll
        for (int r = 0; r < 8; ++r) {
            float m2 = __shfl_xor(m[r], off, 32);
            float s2 = __shfl_xor(s[r], off, 32);
            olmerge(m[r], s[r], m2, s2);
        }
    }

    __shared__ float wm[4][16];
    __shared__ float wsv[4][16];
    if ((L & 15) == 0) {
        int half = L >> 4;
#pragma unroll
        for (int r = 0; r < 8; ++r) {
            wm[wave][half * 8 + r]  = m[r];
            wsv[wave][half * 8 + r] = s[r];
        }
    }
    __syncthreads();

    if (tid < 16) {
        float M = wm[0][tid], S = wsv[0][tid];
#pragma unroll
        for (int w = 1; w < 4; ++w) olmerge(M, S, wm[w][tid], wsv[w][tid]);
        partial[(size_t)blockIdx.x * 32 + tid]      = M;
        partial[(size_t)blockIdx.x * 32 + 16 + tid] = S;
    }
}

// ---------------------------------------------------------------------------
// Kernel 3: fold the NSPLIT partials per row -> stats[row] = (max, 1/sum)
// ---------------------------------------------------------------------------
__global__ void __launch_bounds__(256)
merge_stats(const float* __restrict__ partial, float* __restrict__ stats) {
    const int row = blockIdx.x * blockDim.x + threadIdx.x;
    if (row >= BATCH) return;
    const int rt = row >> 4, r = row & 15;
    float M = -1e30f, S = 0.f;
#pragma unroll
    for (int sp = 0; sp < NSPLIT; ++sp) {
        const float* p = partial + (size_t)(rt * NSPLIT + sp) * 32;
        olmerge(M, S, p[r], p[16 + r]);
    }
    stats[row * 2]     = M;
    stats[row * 2 + 1] = 1.0f / S;
}

// ---------------------------------------------------------------------------
// Kernel 4 (pass 2): recompute logits, emit softmax probabilities with
// nontemporal stores (819 MB stream must not evict L2-resident packB).
// ---------------------------------------------------------------------------
__global__ void __launch_bounds__(128)
pass2_softmax(const unsigned short* __restrict__ proj,
              const unsigned short* __restrict__ packB,
              const float* __restrict__ stats,
              float* __restrict__ out) {
    const int tid  = threadIdx.x;
    const int wave = tid >> 5;
    const int L    = tid & 31;
    const int rt   = blockIdx.x >> 3;
    const int sp   = blockIdx.x & (NSPLIT - 1);
    const int rowbase = rt * 16;
    const int half = L >> 4;
    const int col  = L & 15;

    v16bf a[4];
#pragma unroll
    for (int c = 0; c < 4; ++c) a[c] = load_a_frag(proj, rowbase, L, c);

    float m[8], rinv[8];
#pragma unroll
    for (int r = 0; r < 8; ++r) {
        int row = rowbase + half * 8 + r;
        m[r]    = stats[row * 2];
        rinv[r] = stats[row * 2 + 1];
    }

    for (int nt = sp * 4 + wave; nt < NT; nt += NSPLIT * 4) {
        v8f cf = logits_tile(a, packB, nt, L);
        const int n = nt * 16 + col;
#pragma unroll
        for (int r = 0; r < 8; ++r) {
            int row = rowbase + half * 8 + r;
            float p = __expf(cf[r] - m[r]) * rinv[r];
            __builtin_nontemporal_store(p, out + (size_t)row * VOCAB + n);
        }
    }
}

// ---------------------------------------------------------------------------
// launch
// ---------------------------------------------------------------------------
extern "C" void kernel_launch(void* const* d_in, const int* in_sizes, int n_in,
                              void* d_out, int out_size, void* d_ws, size_t ws_size,
                              hipStream_t stream) {
    const int*   ctx   = (const int*)d_in[0];    // [BATCH, CTX]
    const float* W_in  = (const float*)d_in[1];  // [VOCAB, DIM]
    const float* W_out = (const float*)d_in[2];  // [DIM, VOCAB]
    float*       out   = (float*)d_out;          // [BATCH, VOCAB]

    // workspace carve-out (~26.3 MB total)
    char* ws = (char*)d_ws;
    unsigned short* packB   = (unsigned short*)(ws);                       // 25,600,000 B
    unsigned short* proj    = (unsigned short*)(ws + 25600000);            //    524,288 B
    float*          partial = (float*)(ws + 25600000 + 524288);            //    131,072 B
    float*          stats   = (float*)(ws + 25600000 + 524288 + 131072);   //     16,384 B

    pack_wout   <<<NT,          128, 0, stream>>>(W_out, packB);
    gather_mean <<<BATCH,       128, 0, stream>>>(ctx, W_in, proj);
    pass1_stats <<<RT * NSPLIT, 128, 0, stream>>>(proj, packB, partial);
    merge_stats <<<(BATCH + 255) / 256, 256, 0, stream>>>(partial, stats);
    pass2_softmax<<<RT * NSPLIT, 128, 0, stream>>>(proj, packB, stats, out);
}
